// DC_SpatialAttention_1047972020325
// MI455X (gfx1250) — compile-verified, matrix operand order checked by reading, not run on hardware
//
#include <hip/hip_runtime.h>

typedef __attribute__((ext_vector_type(16))) _Float16 v16h;
typedef __attribute__((ext_vector_type(8)))  float    v8f;

#define B_    16
#define C_    256
#define H_    128
#define W_    128
#define HW_   (H_*W_)
#define K_    7
#define K2_   49
#define KK_   98        // im2col depth = 2*49
#define NOUT_ 147       // 3*49 conv output channels
#define NTIL_ 10        // ceil(147/16) N tiles
#define BN_EPS_ 1e-5f

// ---------------- Kernel 0: zero BN accumulators (re-zeroed every launch) ----
__global__ void k0_zero(float* __restrict__ acc) {
  if (threadIdx.x < 2) acc[threadIdx.x] = 0.f;
}

// ---------------- Kernel 1: per-pixel mean/max over 256 channels ------------
__global__ __launch_bounds__(256) void k1_reduce_xc(const float* __restrict__ x,
                                                    float* __restrict__ xc) {
  int idx = blockIdx.x * 256 + threadIdx.x;          // b*HW + pix
  int b = idx / HW_, pix = idx - b * HW_;
  const float* p = x + (size_t)b * C_ * HW_ + pix;
  float s = 0.f, m = -3.402823466e38f;
  #pragma unroll 4
  for (int c = 0; c < C_; ++c) {
    float v = p[(size_t)c * HW_];
    s += v; m = fmaxf(m, v);
  }
  xc[(size_t)(b*2+0)*HW_ + pix] = s * (1.0f / C_);
  xc[(size_t)(b*2+1)*HW_ + pix] = m;
}

// ---------------- Kernel 2: 7x7 conv (2->147) as WMMA GEMM ------------------
// GEMM: M = B*HW (pixels), N = 147 (pad 160), K = 98 (pad 128, 4 chunks of 32)
// One wave per 16x16 output tile; A = im2col gather of L2-resident xc.
__global__ __launch_bounds__(256) void k2_conv_wmma(const float* __restrict__ xc,
                                                    const float* __restrict__ w_off,
                                                    const float* __restrict__ b_off,
                                                    _Float16* __restrict__ om) {
  const int lane    = threadIdx.x & 31;
  const int wave    = threadIdx.x >> 5;
  const int job     = blockIdx.x * 8 + wave;         // exact: no tail waves
  const int mTile   = job / NTIL_;
  const int nTile   = job - mTile * NTIL_;
  const int halfSel = lane >> 4;                     // K-half select (A/B)
  const int mRow    = lane & 15;                     // A row owned by lane
  const int nCol    = lane & 15;                     // B col owned by lane

  // pixel for this lane's A row (16-pixel tile never crosses batch/row bounds)
  const int gm  = mTile * 16 + mRow;                 // global M = b*HW + pix
  const int b   = gm / HW_;
  const int pix = gm - b * HW_;
  const int y   = pix >> 7;
  const int x   = pix & (W_ - 1);
  const float* xcb = xc + (size_t)b * 2 * HW_;

  const int gn = nTile * 16 + nCol;                  // output channel of B col

  v8f acc = {};
  #pragma unroll
  for (int q = 0; q < 4; ++q) {                      // K chunks of 32
    v16h a, bw;
    #pragma unroll
    for (int j = 0; j < 16; ++j) {
      // ISA 16-bit A layout: lanes 0-15 hold K={0..7,16..23}+32q,
      // lanes 16-31 hold K={8..15,24..31}+32q for the same M rows.
      int k = q * 32 + (j < 8 ? j : j + 8) + 8 * halfSel;
      float av = 0.f, bv = 0.f;
      if (k < KK_) {
        int c  = (k >= K2_) ? 1 : 0;
        int t  = k - c * K2_;
        int ky = t / K_;
        int kx = t - ky * K_;
        int sy = y - 3 + ky, sx = x - 3 + kx;
        if (sy >= 0 && sy < H_ && sx >= 0 && sx < W_)
          av = xcb[c * HW_ + sy * W_ + sx];
        if (gn < NOUT_) bv = w_off[gn * KK_ + k];    // w_off[n][c][ky][kx]
      }
      a[j]  = (_Float16)av;
      bw[j] = (_Float16)bv;
    }
    acc = __builtin_amdgcn_wmma_f32_16x16x32_f16(false, a, false, bw,
                                                 (short)0, acc, false, false);
  }

  if (gn < NOUT_) {
    float bias = b_off[gn];
    #pragma unroll
    for (int v = 0; v < 8; ++v) {
      // D layout: lane&15 = N col; VGPR v -> M row = 8*halfSel + v
      int g  = mTile * 16 + 8 * halfSel + v;
      int bb = g / HW_;
      int pp = g - bb * HW_;
      om[((size_t)bb * NOUT_ + gn) * HW_ + pp] = (_Float16)(acc[v] + bias);
    }
  }
}

// ---------------- bilinear sample with zero padding -------------------------
__device__ __forceinline__ float bilin(const float* __restrict__ img,
                                       int y0, int x0, float dy, float dx) {
  float v00 = (y0   >= 0 && y0   < H_ && x0   >= 0 && x0   < W_) ? img[y0*W_ + x0]       : 0.f;
  float v01 = (y0   >= 0 && y0   < H_ && x0+1 >= 0 && x0+1 < W_) ? img[y0*W_ + x0 + 1]   : 0.f;
  float v10 = (y0+1 >= 0 && y0+1 < H_ && x0   >= 0 && x0   < W_) ? img[(y0+1)*W_ + x0]   : 0.f;
  float v11 = (y0+1 >= 0 && y0+1 < H_ && x0+1 >= 0 && x0+1 < W_) ? img[(y0+1)*W_ + x0+1] : 0.f;
  float wy0 = 1.f - dy, wx0 = 1.f - dx;
  return v00*wy0*wx0 + v01*wy0*dx + v10*dy*wx0 + v11*dy*dx;
}

// ---------------- Kernel 3: deformable sampling + masked dot + BN sums ------
__global__ __launch_bounds__(256) void k3_sample(const float* __restrict__ xc,
                                                 const _Float16* __restrict__ om,
                                                 const float* __restrict__ w_dcn,
                                                 float* __restrict__ out_raw,
                                                 float* __restrict__ acc) {
  __shared__ float wd[KK_];
  if (threadIdx.x < KK_) wd[threadIdx.x] = w_dcn[threadIdx.x];
  __syncthreads();

  int idx = blockIdx.x * 256 + threadIdx.x;          // exact multiple, no tail
  int b = idx / HW_, pix = idx - b * HW_;
  int y = pix >> 7, x = pix & (W_ - 1);
  const float* img0 = xc + (size_t)(b*2+0) * HW_;
  const float* img1 = xc + (size_t)(b*2+1) * HW_;
  const _Float16* omb = om + (size_t)b * NOUT_ * HW_ + pix;

  float out = 0.f;
  #pragma unroll 1
  for (int k = 0; k < K2_; ++k) {
    float oy = (float)omb[(size_t)(2*k    ) * HW_];
    float ox = (float)omb[(size_t)(2*k + 1) * HW_];
    float mv = (float)omb[(size_t)(KK_ + k) * HW_];
    float mk = 1.f / (1.f + __expf(-mv));            // sigmoid mask
    int ky = k / K_, kx = k - ky * K_;
    float ys = (float)(y - 3 + ky) + oy;
    float xs = (float)(x - 3 + kx) + ox;
    float y0f = floorf(ys), x0f = floorf(xs);
    float dy = ys - y0f,  dx = xs - x0f;
    int y0 = (int)y0f, x0 = (int)x0f;
    float s0 = bilin(img0, y0, x0, dy, dx);
    float s1 = bilin(img1, y0, x0, dy, dx);
    out += mk * (s0 * wd[k] + s1 * wd[K2_ + k]);
  }
  out_raw[idx] = out;

  // wave32 reduction, one atomic per wave for BN mean/var sums
  float s = out, s2 = out * out;
  #pragma unroll
  for (int o = 16; o > 0; o >>= 1) {
    s  += __shfl_down(s,  o, 32);
    s2 += __shfl_down(s2, o, 32);
  }
  if ((threadIdx.x & 31) == 0) {
    atomicAdd(&acc[0], s);
    atomicAdd(&acc[1], s2);
  }
}

// ---------------- Kernel 4: batchnorm + affine + sigmoid --------------------
__global__ __launch_bounds__(256) void k4_bn(const float* __restrict__ out_raw,
                                             const float* __restrict__ acc,
                                             const float* __restrict__ gamma,
                                             const float* __restrict__ beta,
                                             float* __restrict__ out) {
  int idx = blockIdx.x * 256 + threadIdx.x;
  const float invN = 1.0f / (float)(B_ * HW_);
  float mean = acc[0] * invN;
  float var  = acc[1] * invN - mean * mean;
  float v = (out_raw[idx] - mean) * rsqrtf(var + BN_EPS_);
  v = v * gamma[0] + beta[0];
  out[idx] = 1.f / (1.f + __expf(-v));
}

extern "C" void kernel_launch(void* const* d_in, const int* in_sizes, int n_in,
                              void* d_out, int out_size, void* d_ws, size_t ws_size,
                              hipStream_t stream) {
  const float* x     = (const float*)d_in[0];   // [16,256,128,128]
  const float* w_off = (const float*)d_in[1];   // [147,2,7,7]
  const float* b_off = (const float*)d_in[2];   // [147]
  const float* w_dcn = (const float*)d_in[3];   // [1,2,7,7] -> 98 floats
  const float* gamma = (const float*)d_in[4];   // [1]
  const float* beta  = (const float*)d_in[5];   // [1]

  char* ws = (char*)d_ws;
  size_t off = 0;
  float*    xc      = (float*)   (ws + off); off += (size_t)B_*2*HW_*sizeof(float);       // 2 MB
  _Float16* om      = (_Float16*)(ws + off); off += (size_t)B_*NOUT_*HW_*sizeof(_Float16);// 73.5 MB
  float*    out_raw = (float*)   (ws + off); off += (size_t)B_*HW_*sizeof(float);         // 1 MB
  float*    acc     = (float*)   (ws + off); off += 2*sizeof(float);

  const int npix = B_ * HW_;                                 // 262144
  k0_zero   <<<1, 32, 0, stream>>>(acc);
  k1_reduce_xc<<<npix / 256, 256, 0, stream>>>(x, xc);
  // wave jobs = (B*HW/16) mTiles * 10 nTiles = 163840 = 20480 blocks * 8 waves
  k2_conv_wmma<<<(npix / 16) * NTIL_ / 8, 256, 0, stream>>>(xc, w_off, b_off, om);
  k3_sample <<<npix / 256, 256, 0, stream>>>(xc, om, w_dcn, out_raw, acc);
  k4_bn     <<<npix / 256, 256, 0, stream>>>(out_raw, acc, gamma, beta, (float*)d_out);
}
